// SocialPooling_69355131895928
// MI455X (gfx1250) — compile-verified, matrix operand order checked by reading, not run on hardware
//
#include <hip/hip_runtime.h>
#include <hip/hip_bf16.h>

// ---------------------------------------------------------------------------
// Social pooling + WMMA bf16 GEMM for MI455X (gfx1250, wave32).
//
// Phase 1: convert/transpose W (f32 [8192,128]) -> Wt bf16 [128,8192]   (ws)
// Phase 2: per-pedestrian LDS max-pool grid (ds_max_u32), A bf16 rows   (ws)
// Phase 3: C[1024,128] = A @ Wt^T + b  via v_wmma_f32_16x16x32_bf16,
//          A tiles staged into LDS by the Tensor Data Mover (TENSORcnt),
//          double-buffered; fragments read back with ds_load_b128.
// ---------------------------------------------------------------------------

typedef __bf16        v16bf __attribute__((ext_vector_type(16)));
typedef __bf16        v8bf  __attribute__((ext_vector_type(8)));
typedef float         v8f   __attribute__((ext_vector_type(8)));
typedef unsigned int  u32x4 __attribute__((ext_vector_type(4)));
typedef int           i32x4 __attribute__((ext_vector_type(4)));
typedef int           i32x8 __attribute__((ext_vector_type(8)));

#define N_PED   1024
#define HID     128
#define GRID_G  8
#define CELLS   (GRID_G * GRID_G)      // 64
#define KDIM    (CELLS * HID)          // 8192
#define KC      512                    // K-chunk staged in LDS (16x512 bf16 = 16 KB)

#if defined(__has_builtin)
#if __has_builtin(__builtin_amdgcn_tensor_load_to_lds)
#define HAVE_TDM 1
#endif
#endif

#if defined(__has_include)
#if __has_include(<hip/amd_detail/amd_gfx1250_TDM.h>)
#define TDM_ARITY6 1   // amdgpu-toolchain (therock headers): 6-arg builtin
#endif
#endif

__device__ __forceinline__ unsigned short f32_to_bf16_rne(float f) {
    unsigned u = __float_as_uint(f);
    unsigned r = (u >> 16) & 1u;
    u += 0x7FFFu + r;
    return (unsigned short)(u >> 16);
}

__device__ __forceinline__ void wait_tensorcnt0() {
#if defined(__has_builtin) && __has_builtin(__builtin_amdgcn_s_wait_tensorcnt)
    __builtin_amdgcn_s_wait_tensorcnt(0);
#else
    asm volatile("s_wait_tensorcnt 0x0" ::: "memory");
#endif
}

#if defined(HAVE_TDM)
// TDM descriptor (ISA 8.3/8.4): 2-D tensor of 2-byte elements, row length
// 8192, row stride 8192; tile = KC x 16 rows, compact row-major into LDS.
__device__ __forceinline__ void tdm_load_a_tile(const unsigned short* gptr,
                                                unsigned ldsByteOff) {
    unsigned long long ga = (unsigned long long)(size_t)gptr;
    u32x4 g0;
    g0[0] = 1u;                                   // count=1, user descriptor
    g0[1] = ldsByteOff;                           // lds_addr [63:32]
    g0[2] = (unsigned)(ga & 0xffffffffull);       // global_addr lo
    g0[3] = (unsigned)((ga >> 32) & 0x01ffffffull) | (2u << 30); // hi | type=2
    i32x8 g1;
    g1[0] = 0x00010000;            // workgroup_mask=0, data_size=1 (2 bytes)
    g1[1] = (int)(0x2000u << 16);  // tensor_dim0 = 8192 (bits [79:48], lo16)
    g1[2] = (int)(0x0400u << 16);  // tensor_dim1 = 1024 (bits [111:80], lo16)
    g1[3] = (int)(((unsigned)KC) << 16);  // tile_dim0 = KC  (bits [127:112])
    g1[4] = 16;                    // tile_dim1 = 16, tile_dim2 = 0
    g1[5] = 8192;                  // tensor_dim0_stride lo32
    g1[6] = 0;                     // stride hi, dim1_stride lo
    g1[7] = 0;
    i32x4 z4 = {0, 0, 0, 0};
#if defined(TDM_ARITY6)
    i32x8 z8 = {0, 0, 0, 0, 0, 0, 0, 0};
    __builtin_amdgcn_tensor_load_to_lds(g0, g1, z4, z4, z8, 0);
#else
    __builtin_amdgcn_tensor_load_to_lds(g0, g1, z4, z4, 0);
#endif
}
#endif // HAVE_TDM

// ---------------------------------------------------------------------------
// Kernel 1: Wt[n][k] = bf16(W[k][n]);  packed uint pairs along k.
// ---------------------------------------------------------------------------
__global__ void sp_convert_w(const float* __restrict__ W,
                             unsigned* __restrict__ WtPacked) {
    int o = blockIdx.x * blockDim.x + threadIdx.x;          // [0, 524288)
    if (o >= HID * (KDIM / 2)) return;
    int n  = o / (KDIM / 2);
    int kp = (o - n * (KDIM / 2)) * 2;
    unsigned lo = f32_to_bf16_rne(W[(size_t)kp * HID + n]);
    unsigned hi = f32_to_bf16_rne(W[(size_t)(kp + 1) * HID + n]);
    WtPacked[o] = lo | (hi << 16);
}

// ---------------------------------------------------------------------------
// Kernel 2: pooling. One block (256 thr = 8 waves) per pedestrian i.
// Non-negative floats order like their uint bit patterns -> ds_max_u32.
// ---------------------------------------------------------------------------
__global__ void __launch_bounds__(256)
sp_pool(const float* __restrict__ hidden,
        const float* __restrict__ pos,
        unsigned* __restrict__ APacked) {
    __shared__ unsigned sgrid[CELLS * HID];   // 32 KB
    __shared__ float2   spos[N_PED];          // 8 KB

    const int i   = blockIdx.x;
    const int tid = threadIdx.x;

    #pragma unroll
    for (int t = tid; t < CELLS * HID; t += 256) sgrid[t] = 0u;
    for (int t = tid; t < N_PED; t += 256)
        spos[t] = ((const float2*)pos)[t];
    __syncthreads();

    const float px = spos[i].x;
    const float py = spos[i].y;
    const int   ch = tid & (HID - 1);
    const int   s  = tid >> 7;                 // two j-streams

    for (int j = s; j < N_PED; j += 2) {
        if (j == i) continue;
        float dx = spos[j].x - px;
        float dy = spos[j].y - py;
        if (__builtin_fabsf(dx) <= 16.0f && __builtin_fabsf(dy) <= 16.0f) {
            int gx = (int)((dx + 16.0f) * 0.25f); gx = gx > 7 ? 7 : gx;
            int gy = (int)((dy + 16.0f) * 0.25f); gy = gy > 7 ? 7 : gy;
            int cell = gy * GRID_G + gx;
            float v = hidden[(size_t)j * HID + ch];
            v = v > 0.0f ? v : 0.0f;           // pre-clamp: max assoc. with 0
            atomicMax(&sgrid[cell * HID + ch], __float_as_uint(v));
        }
    }
    __syncthreads();

    unsigned* rowOut = APacked + (size_t)i * (KDIM / 2);
    for (int t = tid; t < KDIM / 2; t += 256) {
        unsigned lo = f32_to_bf16_rne(__uint_as_float(sgrid[2 * t]));
        unsigned hi = f32_to_bf16_rne(__uint_as_float(sgrid[2 * t + 1]));
        rowOut[t] = lo | (hi << 16);
    }
}

// ---------------------------------------------------------------------------
// Kernel 3: GEMM.  64 blocks (16-row M tiles) x 8 waves; wave w owns the
// 16-col N tile [16w, 16w+16).  A K-chunks (16 x KC bf16) staged into LDS
// (TDM when available, cooperative copy otherwise), double-buffered.
//
// Fragment layout (ISA 7.12.2, 16-bit A 16x32):
//   lane l<16 : row l,    K = k+0..7  and k+16..23
//   lane l>=16: row l-16, K = k+8..15 and k+24..31     (B mirrored on cols)
// ---------------------------------------------------------------------------
__global__ void __launch_bounds__(256)
sp_gemm(const unsigned short* __restrict__ A,    // bf16 bits [1024][8192]
        const unsigned short* __restrict__ Wt,   // bf16 bits [128][8192]
        const float* __restrict__ bias,
        float* __restrict__ out) {
    __shared__ __attribute__((aligned(16))) unsigned short sA[2 * 16 * KC]; // 32 KB

    const unsigned lane = threadIdx.x & 31u;
    const unsigned wave = threadIdx.x >> 5;      // 0..7 -> N tile
    const int m0 = blockIdx.x * 16;
    const int n0 = (int)wave * 16;
    const int r15   = (int)(lane & 15u);
    const int khalf = (int)(lane >> 4) * 8;      // 0 or 8

    const unsigned short* Atile = A + (size_t)m0 * KDIM;      // 16 rows
    const unsigned short* Bb = Wt + (size_t)(n0 + r15) * KDIM + khalf;

    const int aFragBase = r15 * KC + khalf;      // within one LDS buffer

#if defined(HAVE_TDM)
    // ---- TDM double-buffered pipeline -------------------------------------
    if (wave == 0) {
        tdm_load_a_tile(Atile, (unsigned)(size_t)&sA[0]);
        wait_tensorcnt0();
    }
    __syncthreads();

    v8f acc = {};
    for (int c = 0; c < KDIM / KC; ++c) {
        const int bufBase = (c & 1) * (16 * KC);
        if (wave == 0 && c + 1 < KDIM / KC)
            tdm_load_a_tile(Atile + (size_t)(c + 1) * KC,
                            (unsigned)(size_t)&sA[((c + 1) & 1) * (16 * KC)]);
        const unsigned short* Bc = Bb + c * KC;
        #pragma unroll 4
        for (int kk = 0; kk < KC; kk += 32) {
            v8bf a0 = *(const v8bf*)&sA[bufBase + aFragBase + kk];
            v8bf a1 = *(const v8bf*)&sA[bufBase + aFragBase + kk + 16];
            v8bf b0 = *(const v8bf*)(Bc + kk);
            v8bf b1 = *(const v8bf*)(Bc + kk + 16);
            __builtin_prefetch((const void*)(Bc + kk + 64), 0, 3);
            v16bf av = __builtin_shufflevector(a0, a1,
                0, 1, 2, 3, 4, 5, 6, 7, 8, 9, 10, 11, 12, 13, 14, 15);
            v16bf bv = __builtin_shufflevector(b0, b1,
                0, 1, 2, 3, 4, 5, 6, 7, 8, 9, 10, 11, 12, 13, 14, 15);
            acc = __builtin_amdgcn_wmma_f32_16x16x32_bf16(
                false, av, false, bv, (short)0, acc, false, false);
        }
        if (wave == 0 && c + 1 < KDIM / KC) wait_tensorcnt0();
        __syncthreads();
    }
#else
    // ---- Fallback: cooperative A staging, single buffer -------------------
    v8f acc = {};
    const int tid = (int)threadIdx.x;
    for (int c = 0; c < KDIM / KC; ++c) {
        __syncthreads();
        // 16 rows x KC cols = 1024 uint4; 256 threads x 4 each
        #pragma unroll
        for (int t = tid; t < (16 * KC) / 8; t += 256) {
            int row = t >> 6;            // KC/8 = 64 uint4 per row
            int c16 = t & 63;
            *(uint4*)&sA[row * KC + c16 * 8] =
                *(const uint4*)(Atile + (size_t)row * KDIM + c * KC + c16 * 8);
        }
        __syncthreads();
        const unsigned short* Bc = Bb + c * KC;
        #pragma unroll 4
        for (int kk = 0; kk < KC; kk += 32) {
            v8bf a0 = *(const v8bf*)&sA[aFragBase + kk];
            v8bf a1 = *(const v8bf*)&sA[aFragBase + kk + 16];
            v8bf b0 = *(const v8bf*)(Bc + kk);
            v8bf b1 = *(const v8bf*)(Bc + kk + 16);
            __builtin_prefetch((const void*)(Bc + kk + 64), 0, 3);
            v16bf av = __builtin_shufflevector(a0, a1,
                0, 1, 2, 3, 4, 5, 6, 7, 8, 9, 10, 11, 12, 13, 14, 15);
            v16bf bv = __builtin_shufflevector(b0, b1,
                0, 1, 2, 3, 4, 5, 6, 7, 8, 9, 10, 11, 12, 13, 14, 15);
            acc = __builtin_amdgcn_wmma_f32_16x16x32_bf16(
                false, av, false, bv, (short)0, acc, false, false);
        }
    }
#endif

    // D layout: acc[r] -> lanes 0-15: (M=r, N=lane); lanes 16-31: (M=r+8).
    const int n  = n0 + r15;
    const int mb = m0 + (int)(lane >> 4) * 8;
    const float bv_ = bias[n];
    #pragma unroll
    for (int r = 0; r < 8; ++r)
        out[(size_t)(mb + r) * HID + n] = acc[r] + bv_;
}

// ---------------------------------------------------------------------------
extern "C" void kernel_launch(void* const* d_in, const int* in_sizes, int n_in,
                              void* d_out, int out_size, void* d_ws, size_t ws_size,
                              hipStream_t stream) {
    const float* hidden = (const float*)d_in[0];   // [1024,128]
    const float* pos    = (const float*)d_in[1];   // [1024,2]
    const float* W      = (const float*)d_in[2];   // [8192,128]
    const float* bias   = (const float*)d_in[3];   // [128]
    float*       out    = (float*)d_out;           // [1024,128]

    // workspace layout: A bf16 (16 MB) | Wt bf16 (2 MB)
    unsigned char* ws = (unsigned char*)d_ws;
    unsigned* APacked  = (unsigned*)ws;                               // 1024*4096 u32
    unsigned* WtPacked = (unsigned*)(ws + (size_t)N_PED * KDIM * 2);  // 128*4096 u32

    {
        int total = HID * (KDIM / 2);             // 524288
        sp_convert_w<<<(total + 255) / 256, 256, 0, stream>>>(W, WtPacked);
    }
    sp_pool<<<N_PED, 256, 0, stream>>>(hidden, pos, APacked);
    sp_gemm<<<N_PED / 16, 256, 0, stream>>>((const unsigned short*)APacked,
                                            (const unsigned short*)WtPacked,
                                            bias, out);
}
